// RegionSimilarityAnalysis_76184129896955
// MI455X (gfx1250) — matrix operand order repaired and checked by reading, not compiled, not run on hardware
//
#include <hip/hip_runtime.h>
#include <hip/hip_bf16.h>

// Problem constants (match reference: x = (4, 64, 128, 128) f32, K=7)
#define BB   4
#define CC   64
#define HH   128
#define WW   128
#define KK   7
#define PAD  3
#define HP   (HH + 2*PAD)   // 134
#define WP   (WW + 2*PAD)   // 134
#define EPSF 1e-7f

typedef __attribute__((ext_vector_type(16))) _Float16 v16h;
typedef __attribute__((ext_vector_type(8)))  _Float16 v8h;
typedef __attribute__((ext_vector_type(8)))  float    v8f;
typedef __attribute__((ext_vector_type(4)))  float    v4f;

// ---------------------------------------------------------------------------
// Kernel 1: repack (B,C,H,W) f32 -> padded pixel-major (B,HP,WP,C) f16,
//           plus per-pixel L2 norms (B,HP,WP) f32 (0 in halo).
// ---------------------------------------------------------------------------
__global__ __launch_bounds__(256)
void rsa_prep_kernel(const float* __restrict__ x,
                     _Float16* __restrict__ xT,
                     float* __restrict__ nrm)
{
    const int idx   = blockIdx.x * blockDim.x + threadIdx.x;
    const int total = BB * HP * WP;
    if (idx >= total) return;

    const int pc = idx % WP;
    const int pr = (idx / WP) % HP;
    const int b  = idx / (WP * HP);
    const int r  = pr - PAD;
    const int c  = pc - PAD;
    const bool in = (r >= 0) & (r < HH) & (c >= 0) & (c < WW);
    const int rc = in ? r : 0;
    const int cw = in ? c : 0;

    const float* src = x + (size_t)b * CC * HH * WW + (size_t)rc * WW + cw;

    float s = 0.f;
    v8h vec[CC / 8];
#pragma unroll
    for (int ch = 0; ch < CC; ++ch) {
        float f = in ? src[(size_t)ch * HH * WW] : 0.f;
        s += f * f;
        vec[ch >> 3][ch & 7] = (_Float16)f;
    }

    _Float16* dst = xT + (size_t)idx * CC;
#pragma unroll
    for (int v = 0; v < CC / 8; ++v)
        *(v8h*)(dst + 8 * v) = vec[v];         // 16B stores, 128B/lane contiguous

    nrm[idx] = in ? __builtin_sqrtf(s) : 0.f;
}

// ---------------------------------------------------------------------------
// Kernel 2: one block per (b, h); 8 waves, each wave owns 16 output pixels.
// Band-WMMA: for each vertical offset i, two 16x16 D tiles (each = two
// v_wmma_f32_16x16x32_f16 chained over the channel halves) contain all 7
// horizontal-offset dot products in a 7-wide diagonal band:
//   tile t, D[m][n]  ->  j = n - m + 16*t,  neighbor column = w0 + n + 16*t.
// Extraction is branch-free: invalid entries land in a trash LDS slot.
// ---------------------------------------------------------------------------
__global__ __launch_bounds__(256)
void rsa_simsoftmax_kernel(const _Float16* __restrict__ xT,
                           const float* __restrict__ nrm,
                           float* __restrict__ out)
{
    __shared__ float smem[8][16][52];          // [wave][pixel][kk] (+trash+pad)

    const int bh   = blockIdx.x;
    const int b    = bh >> 7;                  // H == 128
    const int h    = bh & (HH - 1);
    const int lane = threadIdx.x & 31;
    const int wave = threadIdx.x >> 5;
    const int w0   = wave * 16;                // pixel tile base (8 waves * 16 = 128)
    const int half = lane >> 4;                // K-subgroup / M-half selector
    const int n    = lane & 15;                // D column this lane holds

    const size_t rowStride = (size_t)WP * CC;                // f16 per padded row
    const _Float16* bbase  = xT  + (size_t)b * HP * rowStride;
    const float*    nb     = nrm + (size_t)b * HP * WP;

    // ---- A fragments: 16 center pixels x 32 channels, contiguous K order (same as B).
    // lane: M = n; elems 0..15 -> K = 16*half + e
    const _Float16* cpix = bbase + (size_t)(h + PAD) * rowStride
                                 + (size_t)(w0 + PAD + n) * CC;
    v16h a[2];
#pragma unroll
    for (int q = 0; q < 2; ++q) {
        a[q] = *(const v16h*)(cpix + q * 32 + 16 * half);
    }

    // ---- Center norms for this lane's 8 output rows (m = r + 8*half): hoisted.
    const float* ncRow = nb + (h + PAD) * WP + (w0 + PAD);
    float ncv[8];
#pragma unroll
    for (int r = 0; r < 8; ++r)
        ncv[r] = ncRow[r + 8 * half];

#pragma unroll 1
    for (int i = 0; i < KK; ++i) {
        const _Float16* rowi = bbase + (size_t)(h + i) * rowStride;
        const float*    nri  = nb + (h + i) * WP;

        if (i < KK - 1)  // prefetch next shifted row's tile into near caches
            __builtin_prefetch(bbase + (size_t)(h + i + 1) * rowStride
                                     + (size_t)w0 * CC, 0, 3);

#pragma unroll
        for (int tile = 0; tile < 2; ++tile) {
            int col = w0 + 16 * tile + n;             // this lane's B column
            col = col < WP ? col : (WP - 1);          // clamp (junk cols unused)

            // Neighbor norm: column == B column, independent of r. One load.
            const float np = nri[col];

            // ---- B fragment: contiguous K order (same as A), lane holds 16 contiguous
            // channels (16*half .. +16) of column `col` -> one 32B load.
            v8f acc = {};
#pragma unroll
            for (int q = 0; q < 2; ++q) {
                v16h bf = *(const v16h*)(rowi + (size_t)col * CC + q * 32 + 16 * half);
                acc = __builtin_amdgcn_wmma_f32_16x16x32_f16(
                          false, a[q], false, bf, (short)0, acc, false, false);
            }

            // ---- Branch-free band extraction: invalid -> trash slot 49.
#pragma unroll
            for (int r = 0; r < 8; ++r) {
                const int m = r + 8 * half;                    // output pixel in tile
                const int j = n - m + 16 * tile;               // horizontal offset
                const int slot = ((unsigned)j < (unsigned)KK) ? (i * KK + j) : 49;
                const float val =
                    acc[r] * __builtin_amdgcn_rcpf(__builtin_fmaf(ncv[r], np, EPSF));
                smem[wave][m][slot] = val;                     // one ds_store_b32
            }
        }
    }

    __syncthreads();

    // ---- Softmax over 49 window positions; threads 0..127 own one pixel each
    // (waves 4..7 skip uniformly; 32-lane coalesced output stores).
    const int p = threadIdx.x;
    if (p < HH) {
        const float* row = &smem[p >> 4][p & 15][0];   // 208B stride: 16B aligned
        float s[KK * KK];
#pragma unroll
        for (int k4 = 0; k4 < 12; ++k4) {              // 12 x ds_load_b128
            const v4f v = *(const v4f*)(row + 4 * k4);
            s[4 * k4 + 0] = v.x; s[4 * k4 + 1] = v.y;
            s[4 * k4 + 2] = v.z; s[4 * k4 + 3] = v.w;
        }
        s[48] = row[48];

        float mx = -3.402823466e38f;
#pragma unroll
        for (int k = 0; k < KK * KK; ++k) mx = fmaxf(mx, s[k]);
        float sum = 0.f;
#pragma unroll
        for (int k = 0; k < KK * KK; ++k) { s[k] = __expf(s[k] - mx); sum += s[k]; }
        const float inv = __builtin_amdgcn_rcpf(sum);

        float* op = out + (size_t)b * (KK * KK) * (HH * WW) + (size_t)h * WW + p;
#pragma unroll 1
        for (int k = 0; k < KK * KK; ++k)
            op[(size_t)k * (HH * WW)] = s[k] * inv;    // 32-lane coalesced per k
    }
}

// ---------------------------------------------------------------------------
extern "C" void kernel_launch(void* const* d_in, const int* in_sizes, int n_in,
                              void* d_out, int out_size, void* d_ws, size_t ws_size,
                              hipStream_t stream)
{
    const float* x   = (const float*)d_in[0];
    float*       out = (float*)d_out;

    // Workspace layout: [ xT f16 (B,HP,WP,C) | norms f32 (B,HP,WP) ]  (~9.5 MB)
    _Float16* xT = (_Float16*)d_ws;
    const size_t xT_bytes = (size_t)BB * HP * WP * CC * sizeof(_Float16);
    float* nrm = (float*)((char*)d_ws + xT_bytes);

    const int totalPix = BB * HP * WP;
    rsa_prep_kernel<<<(totalPix + 255) / 256, 256, 0, stream>>>(x, xT, nrm);
    rsa_simsoftmax_kernel<<<BB * HH, 256, 0, stream>>>(xT, nrm, out);
}
